// Model_80212809220404
// MI455X (gfx1250) — matrix operand order repaired and checked by reading, not compiled, not run on hardware
//
#include <hip/hip_runtime.h>
#include <stdint.h>

#define NN   8192
#define LL   32
#define EE   128
#define HH   128
#define OUTC 16

typedef __bf16 bf16;
typedef __attribute__((ext_vector_type(16))) __bf16 v16bf;
typedef __attribute__((ext_vector_type(8)))  __bf16 v8bf;
typedef __attribute__((ext_vector_type(4)))  __bf16 bf16x4;
typedef __attribute__((ext_vector_type(8)))  float  v8f;
typedef __attribute__((ext_vector_type(4)))  float  f32x4;
typedef __attribute__((ext_vector_type(4)))  unsigned int u32x4;
typedef __attribute__((ext_vector_type(8)))  int    i32x8;
typedef __attribute__((ext_vector_type(4)))  int    i32x4;

static_assert(sizeof(bf16) == 2, "bf16 must be 2 bytes");

#if defined(__AMDGCN__) && __has_builtin(__builtin_amdgcn_tensor_load_to_lds) && \
    __has_builtin(__builtin_amdgcn_s_wait_tensorcnt)
#define USE_TDM 1
#else
#define USE_TDM 0
#endif

// ---------------- helpers ----------------

__device__ __forceinline__ bf16 f2bf(float f) { return (bf16)f; }  // native v_cvt

__device__ __forceinline__ float sigm(float x) {
  return 1.0f / (1.0f + __expf(-x));
}
__device__ __forceinline__ float tanh_fast(float x) {
  x = fminf(fmaxf(x, -15.0f), 15.0f);
  float e = __expf(2.0f * x);
  return (e - 1.0f) / (e + 1.0f);
}

// Pack 16 f32 -> v16bf with hardware packed converts (v_cvt_pk_bf16_f32).
__device__ __forceinline__ v16bf cvt16(f32x4 x0, f32x4 x1, f32x4 x2, f32x4 x3) {
  union { v16bf v; bf16x4 q[4]; } u;
  u.q[0] = __builtin_convertvector(x0, bf16x4);
  u.q[1] = __builtin_convertvector(x1, bf16x4);
  u.q[2] = __builtin_convertvector(x2, bf16x4);
  u.q[3] = __builtin_convertvector(x3, bf16x4);
  return u.v;
}

// A-fragment (16x32 bf16, M on lanes) from an f32 row, converting on the fly.
// ISA 7.12.2: lanes 0-15 hold K=[0..7]+[16..23], lanes 16-31 K=[8..15]+[24..31].
__device__ __forceinline__ v16bf load_a_f32(const float* __restrict__ row, int k0, int hi) {
  const float* p = row + k0 + hi * 8;
  f32x4 x0 = *(const f32x4*)(p);
  f32x4 x1 = *(const f32x4*)(p + 4);
  f32x4 x2 = *(const f32x4*)(p + 16);
  f32x4 x3 = *(const f32x4*)(p + 20);
  return cvt16(x0, x1, x2, x3);
}

// A-fragment from a bf16 row (global or LDS).
__device__ __forceinline__ v16bf load_a_bf(const bf16* __restrict__ row, int k0, int hi) {
  const bf16* p = row + k0 + hi * 8;
  union { v16bf v; v8bf h[2]; } u;
  u.h[0] = *(const v8bf*)(p);
  u.h[1] = *(const v8bf*)(p + 16);
  return u.v;
}

// B-fragment (32x16 bf16, N on lanes): lane's column row-pointer passed in;
// lanes 0-15 take K=[0..7]+[16..23], lanes 16-31 K=[8..15]+[24..31] (matches A).
__device__ __forceinline__ v16bf load_b_bf(const bf16* __restrict__ colrow, int k0, int hi) {
  const bf16* p = colrow + k0 + hi * 8;
  union { v16bf v; v8bf h[2]; } u;
  u.h[0] = *(const v8bf*)(p);
  u.h[1] = *(const v8bf*)(p + 16);
  return u.v;
}

__device__ __forceinline__ v8f wmma_bf16(v16bf a, v16bf b, v8f c) {
  return __builtin_amdgcn_wmma_f32_16x16x32_bf16(false, a, false, b, (short)0, c,
                                                 false, false);
}

// TDM: DMA a [rows=256] x [k=32] bf16 panel (row stride NN elements) into LDS.
// D# per CDNA5 ISA: group0 = {flags, lds_addr, global_addr,
// type=2}, group1 = {data_size=2B, tensor dims/strides, tile dims}.
// This toolchain exposes the 6-arg builtin (g0, g1, g2, g3, extra, cpol).
__device__ __forceinline__ void tdm_load_panel(const bf16* gsrc, unsigned lds_off) {
#if USE_TDM
  unsigned long long ga = (unsigned long long)(uintptr_t)gsrc;
  u32x4 g0;
  g0[0] = 1u;                                          // count=1, user descriptor
  g0[1] = lds_off;                                     // lds_addr (bytes)
  g0[2] = (unsigned)(ga & 0xFFFFFFFFu);                // global_addr[31:0]
  g0[3] = (unsigned)((ga >> 32) & 0x01FFFFFFu) | (2u << 30);  // addr[56:32] | type=2
  i32x8 g1;
  g1[0] = 0x00010000;                  // wg_mask=0, data_size=1 (2 bytes)
  g1[1] = (int)((NN & 0xFFFF) << 16);  // tensor_dim0[15:0] = 8192
  g1[2] = (int)((NN >> 16) | ((256 & 0xFFFF) << 16));  // dim0 hi | tensor_dim1=256
  g1[3] = (int)(32u << 16);            // tensor_dim1 hi=0 | tile_dim0 = 32
  g1[4] = 256;                         // tile_dim1 = 256, tile_dim2 = 0
  g1[5] = NN;                          // tensor_dim0_stride = 8192 elements
  g1[6] = 0;
  g1[7] = 0;
  i32x4 gz4 = {0, 0, 0, 0};
  i32x8 gz8 = {0, 0, 0, 0, 0, 0, 0, 0};
  __builtin_amdgcn_tensor_load_to_lds(g0, g1, gz4, gz4, gz8, 0);
#else
  (void)gsrc; (void)lds_off;
#endif
}

// ---------------- kernel 0: weight prep (f32 -> bf16, transposes, bias sum) ----

__global__ void prep_kernel(const float* __restrict__ Wih_f, const float* __restrict__ Whh_f,
                            const float* __restrict__ b_ih, const float* __restrict__ b_hh,
                            const float* __restrict__ g1, const float* __restrict__ g2,
                            bf16* __restrict__ Wih, bf16* __restrict__ Whh,
                            float* __restrict__ bsum, bf16* __restrict__ g1t,
                            bf16* __restrict__ g2t) {
  int id = blockIdx.x * blockDim.x + threadIdx.x;
  int stride = gridDim.x * blockDim.x;
  for (int i = id; i < 4 * HH * EE; i += stride) {
    Wih[i] = f2bf(Wih_f[i]);
    Whh[i] = f2bf(Whh_f[i]);
  }
  for (int i = id; i < 4 * HH; i += stride) bsum[i] = b_ih[i] + b_hh[i];
  for (int i = id; i < 2 * HH * HH; i += stride) {         // g1t[n][k] = gc1_W[k][n]
    int n = i / HH, k = i % HH;
    g1t[i] = f2bf(g1[k * (2 * HH) + n]);
  }
  for (int i = id; i < OUTC * 2 * HH; i += stride) {       // g2t[n][k] = gc2_W[k][n]
    int n = i / (2 * HH), k = i % (2 * HH);
    g2t[i] = f2bf(g2[k * OUTC + n]);
  }
}

// ---------------- kernel 1: embedding gather + LSTM (final hidden state) ------
// Block = 4 waves = 16-node tile. h state shared in LDS (bf16), c in registers.
// x/h A-fragments loaded & converted once per step, reused across tiles.

__global__ __launch_bounds__(128) void lstm_kernel(
    const int* __restrict__ tokens, const float* __restrict__ embed,
    const bf16* __restrict__ Wih, const bf16* __restrict__ Whh,
    const float* __restrict__ bsum, bf16* __restrict__ feat) {
  __shared__ bf16 hA[16][EE];

  const int tid = threadIdx.x;
  const int lane = tid & 31, wave = tid >> 5;
  const int hi = lane >> 4, col = lane & 15;
  const int mbase = blockIdx.x * 16;

  const bf16 z = f2bf(0.0f);
  for (int i = tid; i < 16 * EE; i += 128) ((bf16*)hA)[i] = z;
  __syncthreads();

  v8f c[2];
  v8f zero8 = {};
  c[0] = zero8;
  c[1] = zero8;

  float bv[2][4];
#pragma unroll
  for (int j = 0; j < 2; ++j) {
    int jt = wave * 2 + j;
#pragma unroll
    for (int g = 0; g < 4; ++g) bv[j][g] = bsum[g * HH + jt * 16 + col];
  }

  const int node = mbase + col;

  for (int t = 0; t < LL; ++t) {
    const int tok = tokens[node * LL + t];
    const float* xrow = embed + (long)tok * EE;

    v16bf ax[4], ah[4];
#pragma unroll
    for (int q = 0; q < 4; ++q) ax[q] = load_a_f32(xrow, q * 32, hi);
#pragma unroll
    for (int q = 0; q < 4; ++q) ah[q] = load_a_bf(&hA[col][0], q * 32, hi);

    v8f hnew[2];
#pragma unroll
    for (int j = 0; j < 2; ++j) {
      const int jt = wave * 2 + j;
      v8f acc[4];
#pragma unroll
      for (int g = 0; g < 4; ++g) acc[g] = zero8;

#pragma unroll
      for (int q = 0; q < 4; ++q) {
        const int k0 = q * 32;
#pragma unroll
        for (int g = 0; g < 4; ++g) {
          const int gu = g * HH + jt * 16 + col;
          v16bf bx = load_b_bf(Wih + (long)gu * EE, k0, hi);
          acc[g] = wmma_bf16(ax[q], bx, acc[g]);
          v16bf bh = load_b_bf(Whh + (long)gu * HH, k0, hi);
          acc[g] = wmma_bf16(ah[q], bh, acc[g]);
        }
      }

      v8f hv;
#pragma unroll
      for (int r = 0; r < 8; ++r) {
        float iv = sigm(acc[0][r] + bv[j][0]);
        float fv = sigm(acc[1][r] + bv[j][1]);
        float gv = tanh_fast(acc[2][r] + bv[j][2]);
        float ov = sigm(acc[3][r] + bv[j][3]);
        float cv = fv * c[j][r] + iv * gv;
        c[j][r] = cv;
        hv[r] = ov * tanh_fast(cv);
      }
      hnew[j] = hv;
    }

    __syncthreads();
#pragma unroll
    for (int j = 0; j < 2; ++j) {
      const int jt = wave * 2 + j;
#pragma unroll
      for (int r = 0; r < 8; ++r)
        hA[r + hi * 8][jt * 16 + col] = f2bf(hnew[j][r]);
    }
    __syncthreads();
  }

  const int row = tid >> 3, c16 = (tid & 7) * 16;
  *(v16bf*)(feat + (long)(mbase + row) * EE + c16) = *(const v16bf*)(&hA[row][c16]);
}

// ---------------- kernel 2: t1t = (feat @ gc1_W)^T  [256][8192] bf16 ----------

__global__ __launch_bounds__(128) void gemm_feat_gc1(
    const bf16* __restrict__ feat, const bf16* __restrict__ g1t, bf16* __restrict__ t1t) {
  const int tid = threadIdx.x;
  const int lane = tid & 31, wave = tid >> 5;
  const int hi = lane >> 4, col = lane & 15;
  const int mbase = blockIdx.x * 16;

  const bf16* arow = feat + (long)(mbase + col) * EE;
  v8f zero8 = {};
  v8f acc[4];
#pragma unroll
  for (int j = 0; j < 4; ++j) acc[j] = zero8;

#pragma unroll
  for (int k0 = 0; k0 < EE; k0 += 32) {
    v16bf a = load_a_bf(arow, k0, hi);
#pragma unroll
    for (int j = 0; j < 4; ++j) {
      const int n = (wave * 4 + j) * 16 + col;
      v16bf b = load_b_bf(g1t + (long)n * EE, k0, hi);
      acc[j] = wmma_bf16(a, b, acc[j]);
    }
  }
#pragma unroll
  for (int j = 0; j < 4; ++j) {
    const int n = (wave * 4 + j) * 16 + col;
#pragma unroll
    for (int r = 0; r < 8; ++r)
      t1t[(long)n * NN + mbase + hi * 8 + r] = f2bf(acc[j][r]);
  }
}

// ---------------- kernel 3: h1 = relu(adj @ t1 + gc1_b)  bf16 [8192][256] -----
// TDM double-buffers the shared 256x32 B-panel into LDS (one DMA per block per
// K-chunk, overlapped with WMMA on the other buffer, TENSORcnt-tracked).
// Each wave owns one 16-row M-tile; adj rows converted f32->bf16 in registers.

__global__ __launch_bounds__(128) void gemm_adj_t1(
    const float* __restrict__ adj, const bf16* __restrict__ t1t,
    const float* __restrict__ gc1_b, bf16* __restrict__ h1) {
  __shared__ __align__(128) bf16 Bbuf[2][256 * 32];  // 2 x 16 KB

  const int tid = threadIdx.x;
  const int lane = tid & 31, wave = tid >> 5;
  const int hi = lane >> 4, col = lane & 15;
  const int mbase = (blockIdx.x * 4 + wave) * 16;

  const float* arow = adj + (long)(mbase + col) * NN;
  v8f zero8 = {};
  v8f acc[16];
#pragma unroll
  for (int j = 0; j < 16; ++j) acc[j] = zero8;

#if USE_TDM
  if (wave == 0)
    tdm_load_panel(t1t, (unsigned)(uintptr_t)&Bbuf[0][0]);
#endif

  const int NCH = NN / 32;
  for (int kc = 0; kc < NCH; ++kc) {
    const int k0 = kc * 32;
    const int buf = kc & 1;

#if USE_TDM
    if (wave == 0) __builtin_amdgcn_s_wait_tensorcnt(0);
    __syncthreads();  // buf ready; prior reads of buf^1 finished
    if (kc + 1 < NCH && wave == 0)
      tdm_load_panel(t1t + k0 + 32, (unsigned)(uintptr_t)&Bbuf[buf ^ 1][0]);
#else
    // synchronous cooperative staging fallback
    __syncthreads();
    for (int i = 0; i < 8; ++i) {
      int idx = i * 128 + tid;          // 1024 x 16B = 16 KB
      int n = idx >> 2, k = (idx & 3) * 8;
      *(v8bf*)&Bbuf[buf][n * 32 + k] = *(const v8bf*)(t1t + (long)n * NN + k0 + k);
    }
    __syncthreads();
#endif

    __builtin_prefetch(arow + k0 + 1024, 0, 1);  // global_prefetch_b8 on adj stream
    v16bf a = load_a_f32(arow, k0, hi);
#pragma unroll
    for (int j = 0; j < 16; ++j) {
      v16bf b = load_b_bf(&Bbuf[buf][(j * 16 + col) * 32], 0, hi);
      acc[j] = wmma_bf16(a, b, acc[j]);
    }
  }

#pragma unroll
  for (int j = 0; j < 16; ++j) {
    const int n = j * 16 + col;
    const float bb = gc1_b[n];
#pragma unroll
    for (int r = 0; r < 8; ++r) {
      float v = acc[j][r] + bb;
      h1[(long)(mbase + hi * 8 + r) * (2 * HH) + n] = f2bf(fmaxf(v, 0.0f));
    }
  }
}

// ---------------- kernel 4: t2t = (h1 @ gc2_W)^T  [16][8192] bf16 -------------

__global__ __launch_bounds__(128) void gemm_h1_gc2(
    const bf16* __restrict__ h1, const bf16* __restrict__ g2t, bf16* __restrict__ t2t) {
  const int tid = threadIdx.x;
  const int lane = tid & 31, wave = tid >> 5;
  const int hi = lane >> 4, col = lane & 15;
  const int mbase = (blockIdx.x * 4 + wave) * 16;

  const bf16* arow = h1 + (long)(mbase + col) * (2 * HH);
  v8f acc = {};
#pragma unroll
  for (int k0 = 0; k0 < 2 * HH; k0 += 32) {
    v16bf a = load_a_bf(arow, k0, hi);
    v16bf b = load_b_bf(g2t + (long)col * (2 * HH), k0, hi);
    acc = wmma_bf16(a, b, acc);
  }
#pragma unroll
  for (int r = 0; r < 8; ++r)
    t2t[(long)col * NN + mbase + hi * 8 + r] = f2bf(acc[r]);
}

// ---------------- kernel 5: out = log_softmax(adj @ t2 + gc2_b) ---------------
// 4 waves split K, reduce partial C tiles through LDS, softmax epilogue.

__global__ __launch_bounds__(128) void gemm_adj_out(
    const float* __restrict__ adj, const bf16* __restrict__ t2t,
    const float* __restrict__ gc2_b, float* __restrict__ out) {
  __shared__ float part[4][8][32];

  const int tid = threadIdx.x;
  const int lane = tid & 31, wave = tid >> 5;
  const int hi = lane >> 4, col = lane & 15;
  const int mbase = blockIdx.x * 16;

  const float* arow = adj + (long)(mbase + col) * NN;
  v8f acc = {};
  for (int k0 = wave * 32; k0 < NN; k0 += 128) {
    __builtin_prefetch(arow + k0 + 2048, 0, 1);
    v16bf a = load_a_f32(arow, k0, hi);
    v16bf b = load_b_bf(t2t + (long)col * NN, k0, hi);
    acc = wmma_bf16(a, b, acc);
  }
#pragma unroll
  for (int r = 0; r < 8; ++r) part[wave][r][lane] = acc[r];
  __syncthreads();

  if (tid < 16) {
    const int m = tid, r = m & 7, h2 = m >> 3;
    float v[OUTC];
    float mx = -1e30f;
#pragma unroll
    for (int n = 0; n < OUTC; ++n) {
      float s = gc2_b[n];
#pragma unroll
      for (int w = 0; w < 4; ++w) s += part[w][r][h2 * 16 + n];
      v[n] = s;
      mx = fmaxf(mx, s);
    }
    float se = 0.0f;
#pragma unroll
    for (int n = 0; n < OUTC; ++n) se += __expf(v[n] - mx);
    float lse = mx + __logf(se);
#pragma unroll
    for (int n = 0; n < OUTC; ++n) out[(long)(mbase + m) * OUTC + n] = v[n] - lse;
  }
}

// ---------------- host side ---------------------------------------------------

extern "C" void kernel_launch(void* const* d_in, const int* in_sizes, int n_in,
                              void* d_out, int out_size, void* d_ws, size_t ws_size,
                              hipStream_t stream) {
  const int*   tokens = (const int*)d_in[0];
  const float* adj    = (const float*)d_in[1];
  const float* embed  = (const float*)d_in[2];
  const float* W_ih   = (const float*)d_in[3];
  const float* W_hh   = (const float*)d_in[4];
  const float* b_ih   = (const float*)d_in[5];
  const float* b_hh   = (const float*)d_in[6];
  const float* gc1_W  = (const float*)d_in[7];
  const float* gc1_b  = (const float*)d_in[8];
  const float* gc2_W  = (const float*)d_in[9];
  const float* gc2_b  = (const float*)d_in[10];
  float* out = (float*)d_out;

  uint8_t* p = (uint8_t*)d_ws;
  auto carve = [&](size_t bytes) -> void* {
    void* r = (void*)p;
    p += (bytes + 255) & ~(size_t)255;
    return r;
  };
  bf16*  Wih_bf = (bf16*)carve((size_t)4 * HH * EE * 2);
  bf16*  Whh_bf = (bf16*)carve((size_t)4 * HH * HH * 2);
  float* bsum   = (float*)carve((size_t)4 * HH * 4);
  bf16*  g1t    = (bf16*)carve((size_t)2 * HH * EE * 2);
  bf16*  g2t    = (bf16*)carve((size_t)OUTC * 2 * HH * 2);
  bf16*  feat   = (bf16*)carve((size_t)NN * EE * 2);
  bf16*  t1t    = (bf16*)carve((size_t)2 * HH * NN * 2);
  bf16*  h1     = (bf16*)carve((size_t)NN * 2 * HH * 2);
  bf16*  t2t    = (bf16*)carve((size_t)OUTC * NN * 2);

  prep_kernel<<<256, 256, 0, stream>>>(W_ih, W_hh, b_ih, b_hh, gc1_W, gc2_W,
                                       Wih_bf, Whh_bf, bsum, g1t, g2t);
  lstm_kernel<<<NN / 16, 128, 0, stream>>>(tokens, embed, Wih_bf, Whh_bf, bsum, feat);
  gemm_feat_gc1<<<NN / 16, 128, 0, stream>>>(feat, g1t, t1t);
  gemm_adj_t1<<<NN / 64, 128, 0, stream>>>(adj, t1t, gc1_b, h1);
  gemm_h1_gc2<<<NN / 64, 128, 0, stream>>>(h1, g2t, t2t);
  gemm_adj_out<<<NN / 16, 128, 0, stream>>>(adj, t2t, gc2_b, out);
}
